// MultiHeadAttention_83854941487223
// MI455X (gfx1250) — matrix operand order repaired and checked by reading, not compiled, not run on hardware
//
#include <hip/hip_runtime.h>

// ---------------------------------------------------------------------------
// Problem constants (from reference): B=4, S=2048, D=1024, H=16, HD=64
// ---------------------------------------------------------------------------
constexpr int B_ = 4;
constexpr int S_ = 2048;
constexpr int D_ = 1024;
constexpr int H_ = 16;
constexpr int HD_ = 64;
constexpr int E3_ = 3 * HD_;     // 192
constexpr int BS_ = B_ * S_;     // 8192

typedef __attribute__((ext_vector_type(16))) __bf16          bf16x16;
typedef __attribute__((ext_vector_type(16))) unsigned short  u16x16;
typedef __attribute__((ext_vector_type(8)))  unsigned short  u16x8;
typedef __attribute__((ext_vector_type(8)))  float           f32x8;
typedef __attribute__((ext_vector_type(4)))  unsigned int    u32x4;
typedef __attribute__((ext_vector_type(8)))  unsigned int    u32x8;

__device__ __forceinline__ unsigned short f2bf(float f) {
  // round-to-nearest-even fp32 -> bf16
  unsigned int u = __builtin_bit_cast(unsigned int, f);
  u += 0x7FFFu + ((u >> 16) & 1u);
  return (unsigned short)(u >> 16);
}

__device__ __forceinline__ bf16x16 join16(u16x8 lo, u16x8 hi) {
  u16x16 r = __builtin_shufflevector(lo, hi, 0,1,2,3,4,5,6,7,8,9,10,11,12,13,14,15);
  return __builtin_bit_cast(bf16x16, r);
}

// Low 32 bits of a flat pointer into LDS == byte offset in LDS (ISA 10.2:
// flat->LDS aperture mapping truncates to addr[31:0]).
__device__ __forceinline__ unsigned lds_off(const void* p) {
  return (unsigned)(size_t)p;
}

// Async global -> LDS copy of 16 bytes, GVS addressing (saddr base + vaddr
// 32-bit byte offset). Tracked by ASYNCcnt.
__device__ __forceinline__ void async_cp16(void* ldst, const void* gbase, unsigned gByteOff) {
  asm volatile("global_load_async_to_lds_b128 %0, %1, %2"
               :: "v"(lds_off(ldst)), "v"(gByteOff), "s"(gbase)
               : "memory");
}
__device__ __forceinline__ void async_wait0() {
  asm volatile("s_wait_asynccnt 0" ::: "memory");
}

// ---------------------------------------------------------------------------
// Tensor Data Mover: 2D bf16 tile (tileD0 x tileD1) from global to LDS.
// D# group0/group1 built per CDNA5 ISA 8.3/8.4. Issued once per calling wave;
// completion via s_wait_tensorcnt.
// ---------------------------------------------------------------------------
__device__ __forceinline__ void tdm_load_2d_bf16(
    void* ldsDst, const void* gsrc,
    unsigned tileD0, unsigned tileD1,        // tile size (elements)
    unsigned tensorD0, unsigned tensorD1,    // tensor dims for OOB (elements)
    unsigned strideD0) {                     // row stride (elements)
  unsigned long long ga = (unsigned long long)(size_t)gsrc;
  u32x4 g0;
  g0[0] = 1u;                                            // count=1, user desc
  g0[1] = lds_off(ldsDst);                               // lds_addr [63:32]
  g0[2] = (unsigned)ga;                                  // global_addr[31:0]
  g0[3] = (unsigned)((ga >> 32) & 0x1FFFFFFu) | (2u << 30); // va[56:32] | type=2
  u32x8 g1;
  g1[0] = 1u << 16;                                      // data_size=1 (2B)
  g1[1] = (tensorD0 & 0xFFFFu) << 16;                    // tensor_dim0[15:0] @ [63:48]
  g1[2] = (tensorD0 >> 16) | ((tensorD1 & 0xFFFFu) << 16); // dim0 hi | dim1 lo
  g1[3] = (tensorD1 >> 16) | (tileD0 << 16);             // dim1 hi | tile_dim0
  g1[4] = tileD1;                                        // tile_dim1 (tile_dim2=0)
  g1[5] = strideD0;                                      // tensor_dim0_stride lo32
  g1[6] = 0;                                             // stride0 hi | stride1 lo
  g1[7] = 0;
  asm volatile("tensor_load_to_lds %0, %1" :: "s"(g0), "s"(g1) : "memory");
}

// ---------------------------------------------------------------------------
// Pass 0a: fp32 -> bf16 (x), vectorized
// ---------------------------------------------------------------------------
__global__ __launch_bounds__(256) void cvt_bf16_kernel(const float* __restrict__ in,
                                                       unsigned short* __restrict__ out,
                                                       int n4) {
  int i = blockIdx.x * 256 + threadIdx.x;
  if (i < n4) {
    const float4 v = ((const float4*)in)[i];
    unsigned short* o = out + (size_t)i * 4;
    o[0] = f2bf(v.x); o[1] = f2bf(v.y); o[2] = f2bf(v.z); o[3] = f2bf(v.w);
  }
}

// Pass 0b: Wqkv [H,D,192] f32 -> [H,192,D] bf16 (transposed, so GEMM staging
// is a contiguous row copy). Coalesced reads, strided writes; one-time cost.
__global__ __launch_bounds__(256) void cvt_wqkv_kernel(const float* __restrict__ in,
                                                       unsigned short* __restrict__ out) {
  int idx = blockIdx.x * 256 + threadIdx.x;      // exactly H*D*E3 threads
  int h = idx / (D_ * E3_);
  int r = idx % (D_ * E3_);
  int d = r / E3_, c = r % E3_;
  out[((size_t)h * E3_ + c) * D_ + d] = f2bf(in[idx]);
}

// Pass 0c: Wo [D,D] f32 -> [Dout,Din] bf16 (transposed)
__global__ __launch_bounds__(256) void cvt_wo_kernel(const float* __restrict__ in,
                                                     unsigned short* __restrict__ out) {
  int idx = blockIdx.x * 256 + threadIdx.x;      // exactly D*D threads
  int k = idx >> 10, c = idx & (D_ - 1);
  out[(size_t)c * D_ + k] = f2bf(in[idx]);
}

// ---------------------------------------------------------------------------
// Pass 1: per-head QKV projection (bf16 WMMA, f32 acc), double-buffered
// async-DMA staging so the DMA engines overlap the matrix pipes.
//   c<64  -> Qb  [bh][s][e]  row-major, pre-scaled by HD^-0.5
//   c<128 -> Kb  [bh][s][e]  row-major
//   else  -> Vtb [bh][e][s]  transposed (attention B-matrix for P.V)
// ---------------------------------------------------------------------------
__global__ __launch_bounds__(256) void qkv_kernel(
    const unsigned short* __restrict__ xb,     // [B,S,D] bf16
    const unsigned short* __restrict__ wqt,    // [H,192,D] bf16 (pre-transposed)
    const float*          __restrict__ bqkv,   // [H,192] f32
    unsigned short* __restrict__ Qb,
    unsigned short* __restrict__ Kb,
    unsigned short* __restrict__ Vtb) {
  const int bh = blockIdx.x;
  const int b  = bh >> 4, h = bh & 15;
  const int s0 = blockIdx.y * 128;
  const int t  = threadIdx.x;
  const int lane = t & 31, w = t >> 5;

  __shared__ unsigned short lA[2][128 * 32];   // x tile, row-major [m][k]
  __shared__ unsigned short lB[2][192 * 32];   // W^T tile, n-major [c][k]

  f32x8 acc[12] = {};
  const int m = lane & 15, hh = lane >> 4;
  const unsigned short* wbase = wqt + (size_t)h * E3_ * D_;

  auto stage = [&](int buf, int kk) {
#pragma unroll
    for (int i = 0; i < 2; i++) {
      int j = t + i * 256;
      int row = j >> 2;
      int cq  = (j & 3) * 8;
      async_cp16(&lA[buf][row * 32 + cq], xb,
                 (unsigned)(((b * S_ + s0 + row) * D_ + kk + cq) * 2));
    }
#pragma unroll
    for (int i = 0; i < 3; i++) {
      int q = t + i * 256;
      int c = q >> 2;
      int part = (q & 3) * 8;
      async_cp16(&lB[buf][c * 32 + part], wbase,
                 (unsigned)((c * D_ + kk + part) * 2));
    }
  };

  stage(0, 0);                       // prologue fill
  int cur = 0;
  for (int kk = 0; kk < D_; kk += 32, cur ^= 1) {
    async_wait0();                   // my copies for buf[cur] complete
    __syncthreads();                 // everyone's copies visible; prev reads done
    if (kk + 32 < D_) stage(cur ^ 1, kk + 32);  // overlap next DMA with compute

    u16x8 a0 = *(const u16x8*)&lA[cur][(w * 16 + m) * 32 + hh * 8];
    u16x8 a1 = *(const u16x8*)&lA[cur][(w * 16 + m) * 32 + 16 + hh * 8];
    bf16x16 A = join16(a0, a1);
#pragma unroll
    for (int j = 0; j < 12; j++) {
      u16x8 b0 = *(const u16x8*)&lB[cur][(j * 16 + m) * 32 + hh * 8];
      u16x8 b1 = *(const u16x8*)&lB[cur][(j * 16 + m) * 32 + 16 + hh * 8];
      acc[j] = __builtin_amdgcn_wmma_f32_16x16x32_bf16(
          false, A, false, join16(b0, b1), (short)0, acc[j], false, false);
    }
  }

  const float scale = 0.125f;  // 64^-0.5
#pragma unroll
  for (int j = 0; j < 12; j++) {
    int c = j * 16 + m;
    float bias = bqkv[h * E3_ + c];
#pragma unroll
    for (int i = 0; i < 8; i++) {
      int s = s0 + w * 16 + hh * 8 + i;
      float v = acc[j][i] + bias;
      if (c < 64)
        Qb[((size_t)bh * S_ + s) * 64 + c] = f2bf(v * scale);
      else if (c < 128)
        Kb[((size_t)bh * S_ + s) * 64 + (c - 64)] = f2bf(v);
      else
        Vtb[((size_t)bh * 64 + (c - 128)) * S_ + s] = f2bf(v);
    }
  }
}

// ---------------------------------------------------------------------------
// Pass 2: causal flash attention per (b,h). Block: 128 threads (4 waves),
// 64 query rows, 64-key blocks, online softmax. K/V tiles staged by the
// Tensor Data Mover, double-buffered (wave 0 issues block kb+1 while all
// waves compute block kb; TENSORcnt completion).
// ---------------------------------------------------------------------------
__global__ __launch_bounds__(128) void attn_kernel(
    const unsigned short* __restrict__ Qb,    // [BH,S,64] (pre-scaled)
    const unsigned short* __restrict__ Kb,    // [BH,S,64]
    const unsigned short* __restrict__ Vtb,   // [BH,64,S]
    unsigned short* __restrict__ Ob) {        // [B,S,D] heads concatenated
  const int bh = blockIdx.x, b = bh >> 4, h = bh & 15;
  const int qb = blockIdx.y;
  const int q0 = qb * 64;
  const int t = threadIdx.x, lane = t & 31, w = t >> 5;
  const int m = lane & 15, hh = lane >> 4;

  __shared__ unsigned short lK[2][64 * 64];     // [key][kd]
  __shared__ unsigned short lV[2][64 * 64];     // [e][key]
  __shared__ unsigned short lP[4][16 * 64];     // per-wave P tile, row-major

  // Q fragments for this wave's 16 query rows (registers, whole kernel)
  const unsigned short* qrow = &Qb[((size_t)bh * S_ + q0 + w * 16 + m) * 64];
  bf16x16 AQ[2];
#pragma unroll
  for (int k2 = 0; k2 < 2; k2++) {
    u16x8 a0 = *(const u16x8*)&qrow[k2 * 32 + hh * 8];
    u16x8 a1 = *(const u16x8*)&qrow[k2 * 32 + 16 + hh * 8];
    AQ[k2] = join16(a0, a1);
  }

  f32x8 accO[4] = {};
  float rmax[8], rsum[8];
#pragma unroll
  for (int i = 0; i < 8; i++) { rmax[i] = -__builtin_inff(); rsum[i] = 0.0f; }

  auto stage = [&](int buf, int k0) {
    // K block: 64 key-rows x 64 kd, row stride 64 -> lK[key][kd]
    tdm_load_2d_bf16(lK[buf], &Kb[((size_t)bh * S_ + k0) * 64],
                     /*tile*/ 64, 64, /*tensor*/ 64, S_, /*stride*/ 64);
    // V^T block: 64 e-rows x 64 keys, row stride S -> lV[e][key]
    tdm_load_2d_bf16(lV[buf], &Vtb[(size_t)bh * 64 * S_ + k0],
                     /*tile*/ 64, 64, /*tensor*/ S_, 64, /*stride*/ S_);
  };

  if (t < 32) stage(0, 0);           // prologue: wave 0 fills buf 0
  int cur = 0;
  for (int kb = 0; kb <= qb; kb++, cur ^= 1) {
    const int k0 = kb * 64;
    if (t < 32) __builtin_amdgcn_s_wait_tensorcnt(0);
    __syncthreads();                 // tile visible; prev buf reads done
    if (kb < qb && t < 32) stage(cur ^ 1, k0 + 64);  // overlap next TDM

    // scores = Q . K^T (already scaled): 4 key-subtiles x 2 K-halves
    f32x8 sc[4] = {};
#pragma unroll
    for (int j = 0; j < 4; j++) {
#pragma unroll
      for (int k2 = 0; k2 < 2; k2++) {
        u16x8 b0 = *(const u16x8*)&lK[cur][(j * 16 + m) * 64 + k2 * 32 + hh * 8];
        u16x8 b1 = *(const u16x8*)&lK[cur][(j * 16 + m) * 64 + k2 * 32 + 16 + hh * 8];
        sc[j] = __builtin_amdgcn_wmma_f32_16x16x32_bf16(
            false, AQ[k2], false, join16(b0, b1), (short)0, sc[j], false, false);
      }
    }

    // causal mask on the diagonal block
    if (kb == qb) {
#pragma unroll
      for (int j = 0; j < 4; j++) {
        int key = k0 + j * 16 + m;
#pragma unroll
        for (int i = 0; i < 8; i++) {
          int q = q0 + w * 16 + hh * 8 + i;
          if (key > q) sc[j][i] = -__builtin_inff();
        }
      }
    }

    // online softmax: row groups are the 16 lanes sharing (lane>>4)
    float alpha[8];
#pragma unroll
    for (int i = 0; i < 8; i++) {
      float bm = fmaxf(fmaxf(sc[0][i], sc[1][i]), fmaxf(sc[2][i], sc[3][i]));
#pragma unroll
      for (int d = 1; d < 16; d <<= 1) bm = fmaxf(bm, __shfl_xor(bm, d, 16));
      float nm = fmaxf(rmax[i], bm);
      alpha[i] = __expf(rmax[i] - nm);
      rmax[i] = nm;
    }
    float rs[8];
#pragma unroll
    for (int i = 0; i < 8; i++) rs[i] = 0.0f;
#pragma unroll
    for (int j = 0; j < 4; j++) {
#pragma unroll
      for (int i = 0; i < 8; i++) {
        float p = __expf(sc[j][i] - rmax[i]);
        sc[j][i] = p;
        rs[i] += p;
      }
    }
#pragma unroll
    for (int i = 0; i < 8; i++) {
      float s = rs[i];
#pragma unroll
      for (int d = 1; d < 16; d <<= 1) s += __shfl_xor(s, d, 16);
      rsum[i] = rsum[i] * alpha[i] + s;
    }
#pragma unroll
    for (int j = 0; j < 4; j++)
#pragma unroll
      for (int i = 0; i < 8; i++) accO[j][i] *= alpha[i];

    // C-layout -> A-layout for P via per-wave LDS round trip
#pragma unroll
    for (int j = 0; j < 4; j++)
#pragma unroll
      for (int i = 0; i < 8; i++)
        lP[w][(hh * 8 + i) * 64 + j * 16 + m] = f2bf(sc[j][i]);
    asm volatile("s_wait_dscnt 0" ::: "memory");  // intra-wave LDS RAW fence

    bf16x16 AP[2];
#pragma unroll
    for (int k2 = 0; k2 < 2; k2++) {
      u16x8 a0 = *(const u16x8*)&lP[w][m * 64 + k2 * 32 + hh * 8];
      u16x8 a1 = *(const u16x8*)&lP[w][m * 64 + k2 * 32 + 16 + hh * 8];
      AP[k2] = join16(a0, a1);
    }
    // accO += P . V
#pragma unroll
    for (int j = 0; j < 4; j++) {
#pragma unroll
      for (int k2 = 0; k2 < 2; k2++) {
        u16x8 b0 = *(const u16x8*)&lV[cur][(j * 16 + m) * 64 + k2 * 32 + hh * 8];
        u16x8 b1 = *(const u16x8*)&lV[cur][(j * 16 + m) * 64 + k2 * 32 + 16 + hh * 8];
        accO[j] = __builtin_amdgcn_wmma_f32_16x16x32_bf16(
            false, AP[k2], false, join16(b0, b1), (short)0, accO[j], false, false);
      }
    }
  }

  // normalize and store heads concatenated: Ob[b, s, h*64 + e]
#pragma unroll
  for (int j = 0; j < 4; j++) {
#pragma unroll
    for (int i = 0; i < 8; i++) {
      int s = q0 + w * 16 + hh * 8 + i;
      float v = accO[j][i] / rsum[i];
      Ob[((size_t)(b * S_ + s)) * D_ + h * 64 + j * 16 + m] = f2bf(v);
    }
  }
}

// ---------------------------------------------------------------------------
// Pass 3: output projection out[BS,D] = Ob . Wo + bo (fp32 out).
// Block: 256 threads (8 waves), tile 128x128, double-buffered async staging.
// ---------------------------------------------------------------------------
__global__ __launch_bounds__(256) void oproj_kernel(
    const unsigned short* __restrict__ Ob,    // [BS,1024] bf16
    const unsigned short* __restrict__ wot,   // [Dout,Din] bf16 (pre-transposed)
    const float* __restrict__ bo,
    float* __restrict__ out) {
  const int r0 = blockIdx.x * 128;
  const int c0 = blockIdx.y * 128;
  const int t = threadIdx.x, lane = t & 31, w = t >> 5;
  const int m = lane & 15, hh = lane >> 4;

  __shared__ unsigned short lA[2][128 * 32];   // O tile, row-major [m][k]
  __shared__ unsigned short lB[2][128 * 32];   // Wo^T tile, n-major [c][k]

  f32x8 acc[8] = {};

  auto stage = [&](int buf, int kk) {
#pragma unroll
    for (int i = 0; i < 2; i++) {
      int j = t + i * 256;
      int row = j >> 2;
      int cq  = (j & 3) * 8;
      async_cp16(&lA[buf][row * 32 + cq], Ob,
                 (unsigned)(((r0 + row) * D_ + kk + cq) * 2));
    }
#pragma unroll
    for (int i = 0; i < 2; i++) {
      int q = t + i * 256;
      int c = q >> 2;
      int part = (q & 3) * 8;
      async_cp16(&lB[buf][c * 32 + part], wot,
                 (unsigned)(((c0 + c) * D_ + kk + part) * 2));
    }
  };

  stage(0, 0);
  int cur = 0;
  for (int kk = 0; kk < D_; kk += 32, cur ^= 1) {
    async_wait0();
    __syncthreads();
    if (kk + 32 < D_) stage(cur ^ 1, kk + 32);

    u16x8 a0 = *(const u16x8*)&lA[cur][(w * 16 + m) * 32 + hh * 8];
    u16x8 a1 = *(const u16x8*)&lA[cur][(w * 16 + m) * 32 + 16 + hh * 8];
    bf16x16 A = join16(a0, a1);
#pragma unroll
    for (int j = 0; j < 8; j++) {
      u16x8 b0 = *(const u16x8*)&lB[cur][(j * 16 + m) * 32 + hh * 8];
      u16x8 b1 = *(const u16x8*)&lB[cur][(j * 16 + m) * 32 + 16 + hh * 8];
      acc[j] = __builtin_amdgcn_wmma_f32_16x16x32_bf16(
          false, A, false, join16(b0, b1), (short)0, acc[j], false, false);
    }
  }

#pragma unroll
  for (int j = 0; j < 8; j++) {
    int c = c0 + j * 16 + m;
    float bias = bo[c];
#pragma unroll
    for (int i = 0; i < 8; i++) {
      int r = r0 + w * 16 + hh * 8 + i;
      out[(size_t)r * D_ + c] = acc[j][i] + bias;
    }
  }
}

// ---------------------------------------------------------------------------
// Host launcher
// ---------------------------------------------------------------------------
extern "C" void kernel_launch(void* const* d_in, const int* in_sizes, int n_in,
                              void* d_out, int out_size, void* d_ws, size_t ws_size,
                              hipStream_t stream) {
  const float* x    = (const float*)d_in[0];   // [B,S,D]
  const float* Wqkv = (const float*)d_in[1];   // [H,D,192]
  const float* bqkv = (const float*)d_in[2];   // [H,192]
  const float* Wo   = (const float*)d_in[3];   // [D,D]
  const float* bo   = (const float*)d_in[4];   // [D]
  float* out = (float*)d_out;

  // workspace layout (bytes)
  char* ws = (char*)d_ws;
  unsigned short* xb  = (unsigned short*)(ws + 0);           // 16 MB x bf16
  unsigned short* wqt = (unsigned short*)(ws + 16777216);    //  6 MB Wqkv^T bf16
  unsigned short* wot = (unsigned short*)(ws + 23068672);    //  2 MB Wo^T bf16
  unsigned short* Qb  = (unsigned short*)(ws + 25165824);    // 16 MB Q (scaled)
  unsigned short* Kb  = (unsigned short*)(ws + 41943040);    // 16 MB K
  unsigned short* Vtb = (unsigned short*)(ws + 58720256);    // 16 MB V^T
  unsigned short* Ob  = (unsigned short*)(ws + 75497472);    // 16 MB attn out
  (void)in_sizes; (void)n_in; (void)out_size; (void)ws_size;

  // Pass 0: convert (and pre-transpose weights) to bf16
  {
    int n4 = (B_ * S_ * D_) / 4;
    cvt_bf16_kernel<<<(n4 + 255) / 256, 256, 0, stream>>>(x, xb, n4);
    cvt_wqkv_kernel<<<(H_ * D_ * E3_) / 256, 256, 0, stream>>>(Wqkv, wqt);
    cvt_wo_kernel<<<(D_ * D_) / 256, 256, 0, stream>>>(Wo, wot);
  }
  // Pass 1: QKV projection (double-buffered async staging)
  qkv_kernel<<<dim3(B_ * H_, S_ / 128), 256, 0, stream>>>(xb, wqt, bqkv, Qb, Kb, Vtb);
  // Pass 2: causal flash attention (double-buffered TDM staging)
  attn_kernel<<<dim3(B_ * H_, S_ / 64), 128, 0, stream>>>(Qb, Kb, Vtb, Ob);
  // Pass 3: output projection (double-buffered async staging)
  oproj_kernel<<<dim3(BS_ / 128, D_ / 128), 256, 0, stream>>>(Ob, wot, bo, out);
}
